// CrossAttention_13743895347614
// MI455X (gfx1250) — matrix operand order repaired and checked by reading, not compiled, not run on hardware
//
#include <hip/hip_runtime.h>
#include <hip/hip_bf16.h>
#include <stdint.h>

// ---------------------------------------------------------------------------
// CDNA5 (gfx1250) causal cross-attention, bf16 WMMA + pipelined async-to-LDS.
//   D_MODEL=1024, H=16, Dh=64, B=4, S=2048  ->  M = B*S = 8192 rows.
// Pipeline: 3x GEMM (f32 in -> bf16 head-split out; K written pre-transposed),
// flash attention (bf16 WMMA, f32 online softmax, DOUBLE-BUFFERED async K/V),
// 1x GEMM (bf16 in via async staging -> f32 out).
// ---------------------------------------------------------------------------

typedef __attribute__((ext_vector_type(16))) __bf16        v16bf;
typedef __attribute__((ext_vector_type(8)))  float         v8f;
typedef __attribute__((ext_vector_type(4)))  unsigned int  v4u;

#define DMODEL  1024
#define NHEADS  16
#define HEADDIM 64
#define SEQ     2048
#define BATCH   4
#define MROWS   (BATCH * SEQ)   // 8192

__device__ __forceinline__ unsigned short f2bf(float f) {
  unsigned int u = __float_as_uint(f);
  u += 0x7FFFu + ((u >> 16) & 1u);      // round-to-nearest-even truncation
  return (unsigned short)(u >> 16);
}

union Frag { v16bf v; v4u q[2]; };

// A fragment (16x32 bf16, MxK): lane<16 -> row M=lane, K={0..7,16..23};
// lane>=16 -> row M=lane-16, K={8..15,24..31}. Two ds_load_b128.
__device__ __forceinline__ v16bf load_a_frag(const unsigned short* rowBase, int koff) {
  Frag f;
  f.q[0] = *(const v4u*)(rowBase + koff);
  f.q[1] = *(const v4u*)(rowBase + koff + 16);
  return f.v;
}

// B fragment (32x16 bf16, KxN, stored [k][n] with stride ld): lane -> N=lane&15,
// lane<16 K={0..7,16..23}; lane>=16 K={8..15,24..31}.
__device__ __forceinline__ v16bf load_b_frag(const unsigned short* p, int ld) {
  union { v16bf v; unsigned short u[16]; } f;
  const int l = threadIdx.x & 31;
  const int n = l & 15, kh = (l >> 4) * 8;
#pragma unroll
  for (int s = 0; s < 8; ++s) {
    f.u[s]     = p[(kh + s) * ld + n];
    f.u[8 + s] = p[(16 + kh + s) * ld + n];
  }
  return f.v;
}

__device__ __forceinline__ v8f wmma_bf16(v16bf a, v16bf b, v8f c) {
  return __builtin_amdgcn_wmma_f32_16x16x32_bf16(false, a, false, b, (short)0, c, false, false);
}

// --- gfx1250 async global->LDS copies (ASYNCcnt-tracked, no VGPR round-trip) ---
__device__ __forceinline__ void async_b128(unsigned lds_off, const void* g) {
  asm volatile("global_load_async_to_lds_b128 %0, %1, off"
               :: "v"(lds_off), "v"((unsigned long long)(uintptr_t)g) : "memory");
}
__device__ __forceinline__ void async_b64(unsigned lds_off, const void* g) {
  asm volatile("global_load_async_to_lds_b64 %0, %1, off"
               :: "v"(lds_off), "v"((unsigned long long)(uintptr_t)g) : "memory");
}
template <int N>
__device__ __forceinline__ void wait_async() {       // S_WAIT_ASYNCCNT <= N
  asm volatile("s_wait_asynccnt %0" :: "n"(N) : "memory");
}
__device__ __forceinline__ unsigned lds_off(const void* p) {
  return (unsigned)(uintptr_t)p;       // generic LDS pointer: low 32 bits = LDS offset
}

// ---------------------------------------------------------------------------
// Tiled GEMM: Y = X @ W^T.  Workgroup tile 128x128, 8 waves (4Mx2N), wave 32x64.
// MODE 0: X f32, write bf16 head-split  [B,H,S,Dh]           (Q, V projections)
// MODE 1: X f32, write bf16 head-split TRANSPOSED [B,H,Dh,S] (K projection)
// MODE 2: X bf16 (async-staged),        write f32 row-major  (output projection)
// ---------------------------------------------------------------------------
template <int MODE>
__global__ __launch_bounds__(256) void gemm_xwT(const void* __restrict__ Xv,
                                                const float* __restrict__ W,
                                                void* __restrict__ Yv) {
  constexpr int K  = DMODEL;
  constexpr int N  = DMODEL;
  constexpr int XS = 40;    // 80B row stride: 16B aligned, bank-spread
  constexpr int WS = 136;   // 272B row stride: 16B aligned

  __shared__ __align__(16) unsigned short Xl[128 * XS];  // X tile [m][k]
  __shared__ __align__(16) unsigned short Wt[32 * WS];   // W tile TRANSPOSED [k][n]

  const float*          Xf = (const float*)Xv;
  const unsigned short* Xb = (const unsigned short*)Xv;

  const int t    = threadIdx.x;
  const int wid  = t >> 5, lane = t & 31;
  const int wm   = wid & 3, wn = wid >> 2;
  const int lm   = lane & 15, half = lane >> 4;
  const int koff = half ? 8 : 0;
  const int rowBase = blockIdx.y * 128;
  const int colBase = blockIdx.x * 128;

  v8f acc[2][4];
  const v8f zero8 = {0.f, 0.f, 0.f, 0.f, 0.f, 0.f, 0.f, 0.f};
#pragma unroll
  for (int mi = 0; mi < 2; ++mi)
#pragma unroll
    for (int ni = 0; ni < 4; ++ni) acc[mi][ni] = zero8;

  for (int kc = 0; kc < K / 32; ++kc) {
    // --- stage X tile ---
    if constexpr (MODE == 2) {
#pragma unroll
      for (int i = 0; i < 2; ++i) {            // bf16 copy -> async b128
        int c  = i * 256 + t;
        int rr = c >> 2, k8 = (c & 3) * 8;
        async_b128(lds_off(&Xl[rr * XS + k8]),
                   Xb + (size_t)(rowBase + rr) * K + (size_t)kc * 32 + k8);
      }
    } else {
#pragma unroll
      for (int i = 0; i < 16; ++i) {           // f32 -> bf16 convert on the fly
        int e = i * 256 + t;
        int r = e >> 5, k = e & 31;
        Xl[r * XS + k] = f2bf(Xf[(size_t)(rowBase + r) * K + (size_t)kc * 32 + k]);
      }
    }
    // --- stage W tile transposed: Wt[k][n] = W[colBase+n][kc*32+k] ---
#pragma unroll
    for (int i = 0; i < 16; ++i) {
      int e = i * 256 + t;
      int n = e >> 5, k = e & 31;
      Wt[k * WS + n] = f2bf(W[(size_t)(colBase + n) * K + (size_t)kc * 32 + k]);
    }
    if (kc + 1 < K / 32)  // global_prefetch_b8 hint for next W tile
      __builtin_prefetch(W + (size_t)(colBase + (t & 127)) * K + (size_t)(kc + 1) * 32, 0, 1);
    if constexpr (MODE == 2) wait_async<0>();
    __syncthreads();

    v16bf a[2], bfr[4];
#pragma unroll
    for (int mi = 0; mi < 2; ++mi)
      a[mi] = load_a_frag(&Xl[(wm * 32 + mi * 16 + lm) * XS], koff);
#pragma unroll
    for (int ni = 0; ni < 4; ++ni)
      bfr[ni] = load_b_frag(&Wt[wn * 64 + ni * 16], WS);
#pragma unroll
    for (int mi = 0; mi < 2; ++mi)
#pragma unroll
      for (int ni = 0; ni < 4; ++ni)
        acc[mi][ni] = wmma_bf16(a[mi], bfr[ni], acc[mi][ni]);
    __syncthreads();
  }

  // --- epilogue: C layout VGPR r -> (M = r + 8*laneHalf, N = lane&15) ---
#pragma unroll
  for (int mi = 0; mi < 2; ++mi)
#pragma unroll
    for (int ni = 0; ni < 4; ++ni)
#pragma unroll
      for (int r = 0; r < 8; ++r) {
        int row = rowBase + wm * 32 + mi * 16 + r + 8 * half;
        int col = colBase + wn * 64 + ni * 16 + lm;
        float v = acc[mi][ni][r];
        int bb = row >> 11, ss = row & (SEQ - 1);
        int hh = col >> 6,  dd = col & 63;
        size_t bh = (size_t)bb * NHEADS + hh;
        if constexpr (MODE == 0) {
          ((unsigned short*)Yv)[(bh * SEQ + ss) * HEADDIM + dd] = f2bf(v);
        } else if constexpr (MODE == 1) {
          ((unsigned short*)Yv)[(bh * HEADDIM + dd) * SEQ + ss] = f2bf(v);   // K transposed
        } else {
          ((float*)Yv)[(size_t)row * N + col] = v;
        }
      }
}

// ---------------------------------------------------------------------------
// Flash attention (causal). Workgroup = (b, h, 64-row Q tile), 4 waves.
// K pre-transposed in memory ([bh][dh][s]) -> Q/K/V staging are pure bf16
// async copies.  K/V are DOUBLE-BUFFERED: block j+1 streams in via ASYNCcnt
// (6 in-flight ops) while block j is computed; s_wait_asynccnt 6 fences only
// the previous block's loads.
// ---------------------------------------------------------------------------
__global__ __launch_bounds__(128) void flash_attn(const unsigned short* __restrict__ Q,
                                                  const unsigned short* __restrict__ Kt,
                                                  const unsigned short* __restrict__ Vm,
                                                  unsigned short* __restrict__ ctx) {
  constexpr int QS = 72, KTS = 40, VS = 72, PS = 40;  // 16B-aligned strides
  __shared__ __align__(16) unsigned short Ql[64 * QS];          // Q tile [m][dh]
  __shared__ __align__(16) unsigned short Ktl[2][64 * KTS];     // K block [dh][key] x2
  __shared__ __align__(16) unsigned short Vl[2][32 * VS];       // V block [key][dh] x2
  __shared__ __align__(16) unsigned short Pl[4 * 16 * PS];      // per-wave P staging

  const int qt = blockIdx.x, h = blockIdx.y, b = blockIdx.z;
  const int bh = b * NHEADS + h;
  const int qbase = qt * 64;
  const unsigned short* Qp = Q  + ((size_t)bh * SEQ + qbase) * HEADDIM;
  const unsigned short* Kp = Kt + (size_t)bh * HEADDIM * SEQ;   // [dh][s]
  const unsigned short* Vp = Vm + (size_t)bh * SEQ * HEADDIM;   // [s][dh]

  const int t = threadIdx.x;

  // issue the 6 async ops staging K/V block j into buffer j&1
  auto issue_kv = [&](int j) {
    const int kb = j * 32;
    unsigned short* kt = Ktl[j & 1];
    unsigned short* vv = Vl[j & 1];
#pragma unroll
    for (int i = 0; i < 4; ++i) {              // K: [dh 0..63][key 0..31], b64 (4 keys)
      int c  = i * 128 + t;
      int dd = c >> 3, k4 = (c & 7) * 4;
      async_b64(lds_off(&kt[dd * KTS + k4]), Kp + (size_t)dd * SEQ + kb + k4);
    }
#pragma unroll
    for (int i = 0; i < 2; ++i) {              // V: [key 0..31][dh 0..63], b128 (8 dh)
      int c  = i * 128 + t;
      int kk = c >> 3, d8 = (c & 7) * 8;
      async_b128(lds_off(&vv[kk * VS + d8]), Vp + (size_t)(kb + kk) * HEADDIM + d8);
    }
  };

  // --- prologue: stage Q tile + K/V block 0, wait all ---
#pragma unroll
  for (int i = 0; i < 4; ++i) {
    int c  = i * 128 + t;
    int rr = c >> 3, d8 = (c & 7) * 8;
    async_b128(lds_off(&Ql[rr * QS + d8]), Qp + rr * HEADDIM + d8);
  }
  issue_kv(0);
  wait_async<0>();
  __syncthreads();

  const int wid = t >> 5, lane = t & 31;
  const int lm = lane & 15, half = lane >> 4;
  const int koff = half ? 8 : 0;

  v16bf aq[2];
#pragma unroll
  for (int c = 0; c < 2; ++c)
    aq[c] = load_a_frag(&Ql[(wid * 16 + lm) * QS + c * 32], koff);

  const v8f zero8 = {0.f, 0.f, 0.f, 0.f, 0.f, 0.f, 0.f, 0.f};
  v8f o[4];
#pragma unroll
  for (int d = 0; d < 4; ++d) o[d] = zero8;
  float m_[8], l_[8];
#pragma unroll
  for (int r = 0; r < 8; ++r) { m_[r] = -1e30f; l_[r] = 0.f; }

  const int nblk = qt * 2 + 2;                 // causal: keys up to qbase+63
  for (int j = 0; j < nblk; ++j) {
    const int kb = j * 32;
    // prefetch block j+1 into the other buffer; fence only block j's 6 loads
    if (j + 1 < nblk) { issue_kv(j + 1); wait_async<6>(); }
    else              { wait_async<0>(); }
    __syncthreads();                           // (A) block j visible to all waves

    const unsigned short* Ktb = Ktl[j & 1];
    const unsigned short* Vb  = Vl[j & 1];

    // ---- scores: S = Q (16x64) @ K^T (64x32), two 16x16 C tiles ----
    v8f s0 = zero8, s1 = zero8;
#pragma unroll
    for (int c = 0; c < 2; ++c) {
      const unsigned short* kr = &Ktb[(c * 32) * KTS];
      s0 = wmma_bf16(aq[c], load_b_frag(kr, KTS), s0);
      s1 = wmma_bf16(aq[c], load_b_frag(kr + 16, KTS), s1);
    }

    // ---- scale + causal mask + online softmax ----
    float rmax[8];
#pragma unroll
    for (int r = 0; r < 8; ++r) {
      int qrow = qbase + wid * 16 + r + 8 * half;
      float v0 = s0[r] * 0.125f;               // 1/sqrt(64)
      float v1 = s1[r] * 0.125f;
      if (kb + lm      > qrow) v0 = -1e30f;
      if (kb + 16 + lm > qrow) v1 = -1e30f;
      s0[r] = v0; s1[r] = v1;
      rmax[r] = fmaxf(v0, v1);
    }
#pragma unroll
    for (int r = 0; r < 8; ++r)
#pragma unroll
      for (int x = 1; x < 16; x <<= 1) rmax[r] = fmaxf(rmax[r], __shfl_xor(rmax[r], x, 32));

    float scale[8], rs[8];
#pragma unroll
    for (int r = 0; r < 8; ++r) {
      float mn = fmaxf(m_[r], rmax[r]);
      scale[r] = __expf(m_[r] - mn);
      m_[r] = mn;
      float p0 = __expf(s0[r] - mn);
      float p1 = __expf(s1[r] - mn);
      s0[r] = p0; s1[r] = p1;
      rs[r] = p0 + p1;
    }
#pragma unroll
    for (int r = 0; r < 8; ++r)
#pragma unroll
      for (int x = 1; x < 16; x <<= 1) rs[r] += __shfl_xor(rs[r], x, 32);
#pragma unroll
    for (int r = 0; r < 8; ++r) l_[r] = l_[r] * scale[r] + rs[r];
#pragma unroll
    for (int d = 0; d < 4; ++d)
#pragma unroll
      for (int r = 0; r < 8; ++r) o[d][r] *= scale[r];

    // ---- P: C layout -> A layout through per-wave LDS scratch ----
    unsigned short* Pw = &Pl[wid * 16 * PS];
#pragma unroll
    for (int r = 0; r < 8; ++r) {
      int Mloc = r + 8 * half;
      Pw[Mloc * PS + lm]      = f2bf(s0[r]);
      Pw[Mloc * PS + 16 + lm] = f2bf(s1[r]);
    }
    v16bf ap = load_a_frag(&Pw[lm * PS], koff);

    // ---- O += P (16x32) @ V (32x64) ----
#pragma unroll
    for (int d = 0; d < 4; ++d)
      o[d] = wmma_bf16(ap, load_b_frag(&Vb[d * 16], VS), o[d]);

    __syncthreads();                           // (B) all waves done with buffer j&1
  }

  // ---- normalize and write ctx[b][s][h*64+dh] as bf16 ----
  float inv[8];
#pragma unroll
  for (int r = 0; r < 8; ++r) inv[r] = 1.0f / l_[r];
#pragma unroll
  for (int d = 0; d < 4; ++d)
#pragma unroll
    for (int r = 0; r < 8; ++r) {
      int row = qbase + wid * 16 + r + 8 * half;
      int col = h * 64 + d * 16 + lm;
      ctx[((size_t)b * SEQ + row) * DMODEL + col] = f2bf(o[d][r] * inv[r]);
    }
}

// ---------------------------------------------------------------------------
extern "C" void kernel_launch(void* const* d_in, const int* in_sizes, int n_in,
                              void* d_out, int out_size, void* d_ws, size_t ws_size,
                              hipStream_t stream) {
  const float* query     = (const float*)d_in[0];
  const float* key_value = (const float*)d_in[1];
  const float* Wq        = (const float*)d_in[2];
  const float* Wk        = (const float*)d_in[3];
  const float* Wv        = (const float*)d_in[4];
  const float* Wo        = (const float*)d_in[5];
  float*       out       = (float*)d_out;

  // workspace: Q [B,H,S,Dh], Kt [B,H,Dh,S], V [B,H,S,Dh], ctx [B,S,D], all bf16
  const size_t per = (size_t)BATCH * NHEADS * SEQ * HEADDIM;   // 8,388,608 elems
  unsigned short* Qws = (unsigned short*)d_ws;
  unsigned short* Kws = Qws + per;
  unsigned short* Vws = Kws + per;
  unsigned short* Ctx = Vws + per;

  dim3 gGemm(DMODEL / 128, MROWS / 128);   // (8, 64)
  gemm_xwT<0><<<gGemm, 256, 0, stream>>>(query,     Wq, Qws);
  gemm_xwT<1><<<gGemm, 256, 0, stream>>>(key_value, Wk, Kws);   // K transposed
  gemm_xwT<0><<<gGemm, 256, 0, stream>>>(key_value, Wv, Vws);

  flash_attn<<<dim3(SEQ / 64, NHEADS, BATCH), 128, 0, stream>>>(Qws, Kws, Vws, Ctx);

  gemm_xwT<2><<<gGemm, 256, 0, stream>>>(Ctx, Wo, out);
}
